// GraphomerLayer_12919261626675
// MI455X (gfx1250) — matrix operand order repaired and checked by reading, not compiled, not run on hardware
//
#include <hip/hip_runtime.h>
#include <math.h>

typedef __attribute__((ext_vector_type(16))) _Float16 v16h;
typedef __attribute__((ext_vector_type(8)))  _Float16 v8h;
typedef __attribute__((ext_vector_type(8)))  float    v8f;

#define WMMA_F16(a,b,c) __builtin_amdgcn_wmma_f32_16x16x32_f16(false,(a),false,(b),(short)0,(c),false,false)

__device__ __forceinline__ v8h cvt8(const float4 a, const float4 b) {
  v8h h;
  h[0]=(_Float16)a.x; h[1]=(_Float16)a.y; h[2]=(_Float16)a.z; h[3]=(_Float16)a.w;
  h[4]=(_Float16)b.x; h[5]=(_Float16)b.y; h[6]=(_Float16)b.z; h[7]=(_Float16)b.w;
  return h;
}
__device__ __forceinline__ v8h cvt8s(const float4 a, const float4 b, float s) {
  v8h h;
  h[0]=(_Float16)(a.x*s); h[1]=(_Float16)(a.y*s); h[2]=(_Float16)(a.z*s); h[3]=(_Float16)(a.w*s);
  h[4]=(_Float16)(b.x*s); h[5]=(_Float16)(b.y*s); h[6]=(_Float16)(b.z*s); h[7]=(_Float16)(b.w*s);
  return h;
}

// ---------------------------------------------------------------------------
// Fragment-major LDS layouts (wave32 WMMA 16x16x32 f16, ISA 7.12.2):
//  A 16x32 block: lane = (m&15) + 16*(k>>4),      j = (k&15)
//  B 32x16 block: lane = (n&15) + 16*(k>>4),      j = (k&15)
//  C/D          : lane = (n&15) + 16*(m>>3),      vgpr r = (m&7)
// A fragment = 32 contiguous bytes at (block*32 + lane)*16 halves.
// ---------------------------------------------------------------------------

enum { EPI_NONE=0, EPI_BIAS=1, EPI_GC=2, EPI_RES=3, EPI_GELU=4, EPI_MOE=5 };

struct GemmP {
  const float* A; const float* B; float* C;
  int M, N, K;
  int lda, ldb, ldc;
  long sA, sB, sC;
  const float* bias;   // [N]
  const float* res;    // [M,N] (stride ldc, batch stride sC)
  const float* root;   // [N]
  const float* gates;  // [M,8]
  int expert;
};

// Double-buffered 128x128 WMMA GEMM. 256 threads = 8 waves (4m x 2n), each
// wave owns 32x64 (2x4 fragments). One barrier per k-step: next tile's
// global loads are issued into registers while the current LDS buffer feeds
// the WMMAs; staged registers are then stored into the alternate buffer.
template<int EPI, bool BT>
__global__ __launch_bounds__(256)
void gemm_wmma(GemmP p)
{
  __shared__ _Float16 As[2][8*32*16];   // 2 x 8KB
  __shared__ _Float16 Bs[2][8*32*16];   // 2 x 8KB

  const int bz = blockIdx.z;
  const float* A  = p.A + (long)bz * p.sA;
  const float* Bm = p.B + (long)bz * p.sB;
  float*       C  = p.C + (long)bz * p.sC;
  const float* res = p.res ? (p.res + (long)bz * p.sC) : nullptr;

  const int tileN = blockIdx.x * 128;
  const int tileM = blockIdx.y * 128;
  const int tid  = threadIdx.x;
  const int lane = tid & 31;
  const int wid  = tid >> 5;
  const int wm   = wid & 3;
  const int wn   = wid >> 2;

  // per-thread staging coordinates (2 runs of 8 elements each)
  int rA[2], cA[2], slA[2], joA[2];
  int nB[2], cB[2], slB[2], joB[2];
  #pragma unroll
  for (int it=0; it<2; it++) {
    int sid = tid + it*256;
    rA[it] = sid >> 2; cA[it] = (sid & 3) * 8;
    slA[it] = ((rA[it]>>4)<<5) + (rA[it]&15) + ((cA[it]>>4)<<4);
    joA[it] = cA[it] & 15;
    if (BT) { nB[it] = sid >> 2;  cB[it] = (sid & 3) * 8; }
    else    { nB[it] = sid & 127; cB[it] = (sid >> 7) * 8; }
    slB[it] = ((nB[it]>>4)<<5) + (nB[it]&15) + ((cB[it]>>4)<<4);
    joB[it] = cB[it] & 15;
  }

  auto loadA = [&](int k0, v8h out[2]) {
    #pragma unroll
    for (int it=0; it<2; it++) {
      const float* src = A + (long)(tileM + rA[it]) * p.lda + k0 + cA[it];
      const float4* s4 = (const float4*)src;
      out[it] = cvt8(s4[0], s4[1]);
      if (k0 + 64 < p.K) __builtin_prefetch(src + 64, 0, 0);  // global_prefetch_b8
    }
  };
  auto loadB = [&](int k0, v8h out[2]) {
    #pragma unroll
    for (int it=0; it<2; it++) {
      if (BT) {
        const float* src = Bm + (long)(tileN + nB[it]) * p.ldb + k0 + cB[it];
        const float4* s4 = (const float4*)src;
        out[it] = cvt8(s4[0], s4[1]);
        if (k0 + 64 < p.K) __builtin_prefetch(src + 64, 0, 0);
      } else {
        v8h h;
        #pragma unroll
        for (int u=0; u<8; u++)
          h[u] = (_Float16)Bm[(long)(k0 + cB[it] + u) * p.ldb + tileN + nB[it]];
        out[it] = h;
        if (k0 + 64 < p.K)
          __builtin_prefetch(Bm + (long)(k0 + 64 + cB[it]) * p.ldb + tileN + nB[it], 0, 0);
      }
    }
  };
  auto storeA = [&](int buf, const v8h in[2]) {
    #pragma unroll
    for (int it=0; it<2; it++) *(v8h*)(As[buf] + slA[it]*16 + joA[it]) = in[it];
  };
  auto storeB = [&](int buf, const v8h in[2]) {
    #pragma unroll
    for (int it=0; it<2; it++) *(v8h*)(Bs[buf] + slB[it]*16 + joB[it]) = in[it];
  };

  v8f acc[2][4];
  #pragma unroll
  for (int i=0;i<2;i++)
    #pragma unroll
    for (int j=0;j<4;j++)
      #pragma unroll
      for (int r=0;r<8;r++) acc[i][j][r] = 0.f;

  v8h aReg[2], bReg[2];
  loadA(0, aReg); loadB(0, bReg);
  storeA(0, aReg); storeB(0, bReg);

  int buf = 0;
  for (int k0 = 0; k0 < p.K; k0 += 32) {
    __syncthreads();
    const bool more = (k0 + 32) < p.K;
    if (more) { loadA(k0 + 32, aReg); loadB(k0 + 32, bReg); }

    v16h afrag[2], bfrag[4];
    #pragma unroll
    for (int mf=0; mf<2; mf++)
      afrag[mf] = *(const v16h*)(As[buf] + (((wm<<1)+mf)*32 + lane)*16);
    #pragma unroll
    for (int nf=0; nf<4; nf++)
      bfrag[nf] = *(const v16h*)(Bs[buf] + (((wn<<2)+nf)*32 + lane)*16);

    #pragma unroll
    for (int mf=0; mf<2; mf++)
      #pragma unroll
      for (int nf=0; nf<4; nf++)
        acc[mf][nf] = WMMA_F16(afrag[mf], bfrag[nf], acc[mf][nf]);

    if (more) { storeA(buf^1, aReg); storeB(buf^1, bReg); buf ^= 1; }
  }

  // ---- epilogue + store ----
  #pragma unroll
  for (int mf=0; mf<2; mf++) {
    #pragma unroll
    for (int nf=0; nf<4; nf++) {
      int col  = tileN + wn*64 + nf*16 + (lane & 15);
      int rowb = tileM + wm*32 + mf*16 + ((lane>>4)<<3);
      #pragma unroll
      for (int r=0; r<8; r++) {
        int row = rowb + r;
        float v = acc[mf][nf][r];
        if (EPI != EPI_NONE) v += p.bias[col];
        if (EPI == EPI_GC)   v += p.root[col] + res[(long)row * p.ldc + col];
        if (EPI == EPI_RES)  v += res[(long)row * p.ldc + col];
        if (EPI == EPI_GELU) v = 0.5f * v * (1.f + erff(v * 0.70710678118654752f));
        if (EPI == EPI_MOE) {
          float w = p.gates[(long)row * 8 + p.expert];
          C[(long)row * p.ldc + col] += w * v;
        } else {
          C[(long)row * p.ldc + col] = v;
        }
      }
    }
  }
}

// ---------------------------------------------------------------------------
// Flash attention, one workgroup per (b, h, 128 q rows); hd=64, N=1024.
// All tiles staged in fragment-major order; S stored transposed; P written
// by the softmax directly in A-fragment order (8 probs per 16B ds store).
// Dynamic LDS: Q16K K16K V16K Sf64K Pf32K + 1K = 145KB (CDNA5: 320KB/WGP).
// ---------------------------------------------------------------------------
__global__ __launch_bounds__(256)
void flash_attn(const float* __restrict__ qkv, float* __restrict__ o,
                int N, float scale)
{
  extern __shared__ char smem[];
  _Float16* Qf = (_Float16*)smem;          // [rblk8][kc2][lane32][j16]
  _Float16* Kf = Qf + 128*64;              // [nblk8][kc2][lane32][j16]
  _Float16* Vf = Kf + 128*64;              // [nblk4][kc4][lane32][j16]
  float*    Sf = (float*)(Vf + 128*64);    // transposed: [col128][row128]
  _Float16* Pf = (_Float16*)(Sf + 128*128);// [rblk8][kc4][lane32][j16]
  float* alphaS = (float*)(Pf + 128*128);  // 128
  float* linvS  = alphaS + 128;            // 128

  const int qb = blockIdx.x, h = blockIdx.y, b = blockIdx.z;
  const int tid = threadIdx.x, lane = tid & 31, wid = tid >> 5;
  const int ld = 1536;
  const long tBase   = (long)b * N;
  const long qkvBase = tBase * ld;

  // ---- stage Q (pre-scaled), fragment-major A layout ----
  #pragma unroll
  for (int it=0; it<4; it++) {
    int sid = tid + it*256;               // 1024 runs of 8
    int r = sid >> 3, c = (sid & 7) * 8;
    const float4* s4 = (const float4*)(qkv + qkvBase + (long)(qb*128 + r)*ld + h*64 + c);
    v8h hq = cvt8s(s4[0], s4[1], scale);
    int cm = c & 31, kc = c >> 5;
    int ls = (r&15) + (((cm>>4)&1)<<4);
    *(v8h*)(Qf + ((((r>>4)<<1)+kc)*32 + ls)*16 + (cm&15)) = hq;
  }

  float m_run = -INFINITY, l_run = 0.f;    // row owner: tid < 128

  v8f oacc[2][2];
  #pragma unroll
  for (int i=0;i<2;i++)
    #pragma unroll
    for (int j=0;j<2;j++)
      #pragma unroll
      for (int r=0;r<8;r++) oacc[i][j][r] = 0.f;

  const int wmS = wid & 3, wnS = wid >> 2;   // S: 32x64 per wave
  const int wmO = wid & 3, wnO = wid >> 2;   // O: 32x32 per wave

  for (int kb = 0; kb < N/128; kb++) {
    __syncthreads();
    // ---- stage K (B layout for S) ----
    #pragma unroll
    for (int it=0; it<4; it++) {
      int sid = tid + it*256;
      int r = sid >> 3, c = (sid & 7) * 8;
      const float4* s4 = (const float4*)(qkv + qkvBase + (long)(kb*128 + r)*ld + 512 + h*64 + c);
      v8h hk = cvt8(s4[0], s4[1]);
      int kc = c >> 5;
      int ls = (r&15) + ((((c&31)>>4)&1)<<4);
      *(v8h*)(Kf + ((((r>>4)<<1)+kc)*32 + ls)*16 + (c&15)) = hk;
    }
    // ---- stage V (B layout for O): gather 8 rows per run ----
    #pragma unroll
    for (int it=0; it<4; it++) {
      int c = tid & 63;
      int rbase = it*32 + ((tid>>6)<<3);
      v8h hv;
      #pragma unroll
      for (int u=0; u<8; u++)
        hv[u] = (_Float16)qkv[qkvBase + (long)(kb*128 + rbase + u)*ld + 1024 + h*64 + c];
      int ls = (c&15) + ((((rbase&31)>>4)&1)<<4);
      *(v8h*)(Vf + (((c>>4)*4 + (rbase>>5))*32 + ls)*16 + (rbase&15)) = hv;
    }
    __syncthreads();

    // ---- S = Q*K^T (128x128, K=64) ----
    v8f sacc[2][4];
    #pragma unroll
    for (int i=0;i<2;i++)
      #pragma unroll
      for (int j=0;j<4;j++)
        #pragma unroll
        for (int r=0;r<8;r++) sacc[i][j][r] = 0.f;

    #pragma unroll
    for (int kc=0; kc<2; kc++) {
      v16h a[2], bf[4];
      #pragma unroll
      for (int mf=0; mf<2; mf++)
        a[mf] = *(const v16h*)(Qf + ((((wmS*2+mf)<<1)+kc)*32 + lane)*16);
      #pragma unroll
      for (int nf=0; nf<4; nf++)
        bf[nf] = *(const v16h*)(Kf + ((((wnS*4+nf)<<1)+kc)*32 + lane)*16);
      #pragma unroll
      for (int mf=0; mf<2; mf++)
        #pragma unroll
        for (int nf=0; nf<4; nf++)
          sacc[mf][nf] = WMMA_F16(a[mf], bf[nf], sacc[mf][nf]);
    }
    // store S transposed: lane's 8 rows are contiguous floats
    #pragma unroll
    for (int mf=0; mf<2; mf++)
      #pragma unroll
      for (int nf=0; nf<4; nf++) {
        int col  = wnS*64 + nf*16 + (lane & 15);
        int rowb = wmS*32 + mf*16 + ((lane>>4)<<3);
        *(v8f*)(Sf + col*128 + rowb) = sacc[mf][nf];
      }
    __syncthreads();

    // ---- online softmax: thread tid owns q-row tid (tid < 128) ----
    if (tid < 128) {
      float mx = m_run;
      for (int j=0; j<128; j++) mx = fmaxf(mx, Sf[j*128 + tid]);
      float alpha = __expf(m_run - mx);
      float lsum = 0.f;
      for (int jb=0; jb<128; jb+=8) {
        v8h ph;
        #pragma unroll
        for (int u=0; u<8; u++) {
          float pv = __expf(Sf[(jb+u)*128 + tid] - mx);
          lsum += pv; ph[u] = (_Float16)pv;
        }
        int cm = jb & 31, kc = jb >> 5;
        int ls = (tid&15) + (((cm>>4)&1)<<4);
        *(v8h*)(Pf + ((((tid>>4)<<2)+kc)*32 + ls)*16 + (cm&15)) = ph;
      }
      l_run = l_run * alpha + lsum;
      m_run = mx;
      alphaS[tid] = alpha;
    }
    __syncthreads();

    // ---- rescale O, then O += P*V (128x64, K=128) ----
    #pragma unroll
    for (int mf=0; mf<2; mf++) {
      int rowb = wmO*32 + mf*16 + ((lane>>4)<<3);
      #pragma unroll
      for (int r=0;r<8;r++) {
        float al = alphaS[rowb + r];
        #pragma unroll
        for (int nf=0; nf<2; nf++) oacc[mf][nf][r] *= al;
      }
    }
    #pragma unroll
    for (int kc=0; kc<4; kc++) {
      v16h a[2], bf[2];
      #pragma unroll
      for (int mf=0; mf<2; mf++)
        a[mf] = *(const v16h*)(Pf + ((((wmO*2+mf)<<2)+kc)*32 + lane)*16);
      #pragma unroll
      for (int nf=0; nf<2; nf++)
        bf[nf] = *(const v16h*)(Vf + ((((wnO*2+nf)<<2)+kc)*32 + lane)*16);
      #pragma unroll
      for (int mf=0; mf<2; mf++)
        #pragma unroll
        for (int nf=0; nf<2; nf++)
          oacc[mf][nf] = WMMA_F16(a[mf], bf[nf], oacc[mf][nf]);
    }
  }

  __syncthreads();
  if (tid < 128) linvS[tid] = 1.f / l_run;
  __syncthreads();

  #pragma unroll
  for (int mf=0; mf<2; mf++)
    #pragma unroll
    for (int nf=0; nf<2; nf++) {
      int col  = wnO*32 + nf*16 + (lane & 15);
      int rowb = wmO*32 + mf*16 + ((lane>>4)<<3);
      #pragma unroll
      for (int r=0;r<8;r++) {
        int row = rowb + r;
        o[(tBase + qb*128 + row)*512 + h*64 + col] = oacc[mf][nf][r] * linvS[row];
      }
    }
}

// ---------------------------------------------------------------------------
template<bool ADD>
__global__ __launch_bounds__(256)
void layernorm_k(const float* __restrict__ a, const float* __restrict__ b2,
                 const float* __restrict__ g, const float* __restrict__ bi,
                 float* __restrict__ out)
{
  __shared__ float red[256], red2[256];
  const long row = blockIdx.x;
  const int tid = threadIdx.x;
  float vals[2], sum = 0.f, sq = 0.f;
  #pragma unroll
  for (int i=0;i<2;i++) {
    int c = tid + i*256;
    float v = a[row*512 + c];
    if (ADD) v += b2[row*512 + c];
    vals[i] = v; sum += v; sq += v*v;
  }
  red[tid] = sum; red2[tid] = sq;
  __syncthreads();
  for (int s=128; s>0; s>>=1) {
    if (tid < s) { red[tid] += red[tid+s]; red2[tid] += red2[tid+s]; }
    __syncthreads();
  }
  float mu  = red[0] * (1.f/512.f);
  float var = red2[0] * (1.f/512.f) - mu*mu;
  float rs  = rsqrtf(var + 1e-5f);
  #pragma unroll
  for (int i=0;i<2;i++) {
    int c = tid + i*256;
    out[row*512 + c] = (vals[i] - mu) * rs * g[c] + bi[c];
  }
}

// ---------------------------------------------------------------------------
__global__ __launch_bounds__(256)
void gate_k(const float* __restrict__ x, const float* __restrict__ gw,
            float* __restrict__ w)
{
  const int lane = threadIdx.x & 31, wid = threadIdx.x >> 5;
  const long t = (long)blockIdx.x * 8 + wid;
  float xr[16];
  #pragma unroll
  for (int i=0;i<16;i++) xr[i] = x[t*512 + lane + i*32];
  float s[8];
  #pragma unroll
  for (int e=0;e<8;e++) {
    float acc = 0.f;
    #pragma unroll
    for (int i=0;i<16;i++) acc += xr[i] * gw[e*512 + lane + i*32];
    s[e] = acc;
  }
  #pragma unroll
  for (int off=16; off>0; off>>=1)
    #pragma unroll
    for (int e=0;e<8;e++) s[e] += __shfl_xor(s[e], off, 32);
  if (lane == 0) {
    float mx = s[0];
    for (int e=1;e<8;e++) mx = fmaxf(mx, s[e]);
    float p[8], tot = 0.f;
    for (int e=0;e<8;e++) { p[e] = __expf(s[e]-mx); tot += p[e]; }
    for (int e=0;e<8;e++) p[e] /= tot;
    int i1 = 0;
    for (int e=1;e<8;e++) if (p[e] > p[i1]) i1 = e;
    int i2 = (i1 == 0) ? 1 : 0;
    for (int e=0;e<8;e++) { if (e==i1) continue; if (p[e] > p[i2]) i2 = e; }
    float norm = p[i1] + p[i2];
    for (int e=0;e<8;e++) w[t*8+e] = 0.f;
    w[t*8+i1] = p[i1]/norm;
    w[t*8+i2] = p[i2]/norm;
  }
}

__global__ void fill_zero(float* p, long n)
{
  long i = (long)blockIdx.x * blockDim.x + threadIdx.x;
  if (i < n) p[i] = 0.f;
}

// ---------------------------------------------------------------------------
extern "C" void kernel_launch(void* const* d_in, const int* in_sizes, int n_in,
                              void* d_out, int out_size, void* d_ws, size_t ws_size,
                              hipStream_t stream)
{
  (void)in_sizes; (void)n_in; (void)out_size; (void)ws_size;
  const float* x     = (const float*)d_in[0];
  const float* adj   = (const float*)d_in[1];
  const float* gc_w  = (const float*)d_in[2];
  const float* gc_b  = (const float*)d_in[3];
  const float* root  = (const float*)d_in[4];
  const float* in_w  = (const float*)d_in[5];
  const float* in_b  = (const float*)d_in[6];
  const float* out_w = (const float*)d_in[7];
  const float* out_b = (const float*)d_in[8];
  const float* ln1g  = (const float*)d_in[9];
  const float* ln1b  = (const float*)d_in[10];
  const float* ln2g  = (const float*)d_in[11];
  const float* ln2b  = (const float*)d_in[12];
  const float* gatew = (const float*)d_in[13];
  const float* W1    = (const float*)d_in[14];
  const float* b1    = (const float*)d_in[15];
  const float* W2    = (const float*)d_in[16];
  const float* b2    = (const float*)d_in[17];
  float* out = (float*)d_out;

  const int B = 8, N = 1024, D = 512, DFF = 2048, E = 8;
  const long T = (long)B * N, TD = T * D;

  float* w0   = (float*)d_ws;
  float* g0   = w0; w0 += TD;
  float* x1   = w0; w0 += TD;
  float* qkv  = w0; w0 += T * 3 * D;
  float* oat  = w0; w0 += TD;
  float* res1 = w0; w0 += TD;
  float* x2   = w0; w0 += TD;
  float* moe  = w0; w0 += TD;
  float* gts  = w0; w0 += T * E;
  float* hid  = w0; w0 += T * DFF;

  dim3 blk(256);
  GemmP p;

  // 1) g0 = adj @ x (batched over B)
  p = {}; p.A = adj; p.B = x; p.C = g0;
  p.M = N; p.N = D; p.K = N; p.lda = N; p.ldb = D; p.ldc = D;
  p.sA = (long)N*N; p.sB = (long)N*D; p.sC = (long)N*D;
  gemm_wmma<EPI_NONE,false><<<dim3(D/128, N/128, B), blk, 0, stream>>>(p);

  // 2) x1 = g0 @ gc_w^T + gc_b + root + x
  p = {}; p.A = g0; p.B = gc_w; p.C = x1;
  p.M = (int)T; p.N = D; p.K = D; p.lda = D; p.ldb = D; p.ldc = D;
  p.bias = gc_b; p.root = root; p.res = x;
  gemm_wmma<EPI_GC,true><<<dim3(D/128, T/128, 1), blk, 0, stream>>>(p);

  // 3) qkv = x1 @ in_w^T + in_b
  p = {}; p.A = x1; p.B = in_w; p.C = qkv;
  p.M = (int)T; p.N = 3*D; p.K = D; p.lda = D; p.ldb = D; p.ldc = 3*D;
  p.bias = in_b;
  gemm_wmma<EPI_BIAS,true><<<dim3(3*D/128, T/128, 1), blk, 0, stream>>>(p);

  // 4) flash attention -> oat
  size_t shmem = (size_t)3*128*64*2 + 128*128*4 + 128*128*2 + 256*4;
  flash_attn<<<dim3(N/128, 8, B), blk, shmem, stream>>>(qkv, oat, N, 0.125f);

  // 5) res1 = x1 + oat @ out_w^T + out_b
  p = {}; p.A = oat; p.B = out_w; p.C = res1;
  p.M = (int)T; p.N = D; p.K = D; p.lda = D; p.ldb = D; p.ldc = D;
  p.bias = out_b; p.res = x1;
  gemm_wmma<EPI_RES,true><<<dim3(D/128, T/128, 1), blk, 0, stream>>>(p);

  // 6) x2 = LN1(res1)
  layernorm_k<false><<<dim3((unsigned)T), blk, 0, stream>>>(res1, nullptr, ln1g, ln1b, x2);

  // 7) gates
  gate_k<<<dim3((unsigned)(T/8)), blk, 0, stream>>>(x2, gatew, gts);

  // 8) moe = 0
  fill_zero<<<dim3((unsigned)((TD + 255)/256)), blk, 0, stream>>>(moe, TD);

  // 9) per-expert FFN, gate-weighted accumulate
  for (int e = 0; e < E; e++) {
    p = {}; p.A = x2; p.B = W1 + (long)e*D*DFF; p.C = hid;
    p.M = (int)T; p.N = DFF; p.K = D; p.lda = D; p.ldb = DFF; p.ldc = DFF;
    p.bias = b1 + (long)e*DFF;
    gemm_wmma<EPI_GELU,false><<<dim3(DFF/128, T/128, 1), blk, 0, stream>>>(p);

    p = {}; p.A = hid; p.B = W2 + (long)e*DFF*D; p.C = moe;
    p.M = (int)T; p.N = D; p.K = DFF; p.lda = DFF; p.ldb = D; p.ldc = D;
    p.bias = b2 + (long)e*D; p.gates = gts; p.expert = e;
    gemm_wmma<EPI_MOE,false><<<dim3(D/128, T/128, 1), blk, 0, stream>>>(p);
  }

  // 10) out = LN2(x2 + moe)
  layernorm_k<true><<<dim3((unsigned)T), blk, 0, stream>>>(x2, moe, ln2g, ln2b, out);
}
